// CHSHistoryCrossAttentionFusion_8289286881420
// MI455X (gfx1250) — matrix operand order repaired and checked by reading, not compiled, not run on hardware
//
#include <hip/hip_runtime.h>
#include <math.h>

// ---------------------------------------------------------------------------
// CHS history cross-attention fusion for gfx1250 (CDNA5, wave32, WMMA).
// All matmuls: v_wmma_f32_16x16x32_bf16 (bf16 in, f32 accumulate).
// bf16 conversion + weight transpose done once up front; hot GEMM loop is
// double-buffered LDS ping-pong: b128 loads + WMMA, one barrier per K-tile.
// ---------------------------------------------------------------------------

typedef __attribute__((ext_vector_type(16))) __bf16        v16bf;
typedef __attribute__((ext_vector_type(8)))  float         v8f;
typedef __attribute__((ext_vector_type(4)))  unsigned int  v4u;

#define S_LEN   4096
#define D_DIM   1024
#define N_CTX   512
#define RMS_EPS 1e-6f

union FragCvt  { v4u u2[2]; v16bf b; };
union PackBF8  { unsigned short s[8]; v4u u; };

__device__ __forceinline__ unsigned short f2bfu(float f) {
  union { float f; unsigned u; } x; x.f = f;
  unsigned r = x.u + 0x7FFFu + ((x.u >> 16) & 1u);   // round-to-nearest-even
  return (unsigned short)(r >> 16);
}
// sinusoidal position embedding value for feature d (D=1024, half=512)
__device__ __forceinline__ float pe_val(int pos, int d) {
  float invf = expf((float)(d & 511) * (-9.210340371976184f / 512.0f));
  float ang  = (float)pos * invf;
  return (d < 512) ? sinf(ang) : cosf(ang);
}

// ---------------------------------------------------------------------------
// Elementwise f32 -> bf16 (4 per thread)
// ---------------------------------------------------------------------------
__global__ __launch_bounds__(256)
void conv_bf16_kernel(const float* __restrict__ X, unsigned short* __restrict__ Y) {
  size_t i = ((size_t)blockIdx.x * 256 + threadIdx.x) * 4;
  #pragma unroll
  for (int j = 0; j < 4; ++j) Y[i + j] = f2bfu(X[i + j]);
}

// ---------------------------------------------------------------------------
// Weight transpose + convert: W[K,N] f32 -> Wt[N,K] bf16 (32x32 LDS tiles)
// grid = (N/32, K/32), block 256 (32x8)
// ---------------------------------------------------------------------------
__global__ __launch_bounds__(256)
void transpose_conv_kernel(const float* __restrict__ W, unsigned short* __restrict__ Wt,
                           int K, int N) {
  __shared__ float tile[32][33];
  const int n0 = blockIdx.x * 32, k0 = blockIdx.y * 32;
  const int tx = threadIdx.x & 31, ty = threadIdx.x >> 5;   // ty 0..7
  #pragma unroll
  for (int i = 0; i < 32; i += 8)
    tile[ty + i][tx] = W[(size_t)(k0 + ty + i) * N + (n0 + tx)];
  __syncthreads();
  #pragma unroll
  for (int i = 0; i < 32; i += 8)
    Wt[(size_t)(n0 + ty + i) * K + (k0 + tx)] = f2bfu(tile[tx][ty + i]);
}

// ---------------------------------------------------------------------------
// GEMM: C[M,1024] = A[M,K](bf16) @ Bt[1024,K](bf16, pre-transposed)
// Block tile 128x128, 8 waves (4x2), wave tile 32x64, K-step 64.
// Double-buffered LDS ping-pong; one barrier per K-tile.
// out_mode 0: C f32 row-major [M,1024]
// out_mode 1: C bf16 transposed [1024, M]  (packed b128 stores; for V^T)
// ---------------------------------------------------------------------------
__global__ __launch_bounds__(256)
void gemm_bf16_kernel(const unsigned short* __restrict__ A,
                      const unsigned short* __restrict__ Bt,
                      void* __restrict__ Cout, int M, int K, int out_mode) {
  __shared__ __align__(16) unsigned short As[2][128][72];   // 2 x (128x64 + pad)
  __shared__ __align__(16) unsigned short Bs[2][128][72];

  const int tid  = threadIdx.x;
  const int lane = tid & 31;
  const int wid  = tid >> 5;
  const int wm   = wid >> 1;          // 0..3
  const int wn   = wid & 1;           // 0..1
  const int m0   = blockIdx.y * 128;
  const int n0   = blockIdx.x * 128;

  v8f vzero = {0.f,0.f,0.f,0.f,0.f,0.f,0.f,0.f};
  v8f acc[2][4];
  #pragma unroll
  for (int mi = 0; mi < 2; ++mi) {
    #pragma unroll
    for (int ni = 0; ni < 4; ++ni) acc[mi][ni] = vzero;
  }

  const int srow = tid >> 1;            // 0..127 (A row / Bt col)
  const int scol = (tid & 1) * 32;      // 0 or 32 (k offset)
  const int kbA  = (lane < 16) ? 0 : 8;
  const int kbB  = (lane < 16) ? 0 : 16;
  const int hi8  = (lane < 16) ? 0 : 8;

  const unsigned short* gA = A  + (size_t)(m0 + srow) * K + scol;
  const unsigned short* gB = Bt + (size_t)(n0 + srow) * K + scol;

  // prologue: stage K-tile 0 into buffer 0
  {
    const v4u* ga = (const v4u*)gA;
    const v4u* gb = (const v4u*)gB;
    v4u* la = (v4u*)&As[0][srow][scol];
    v4u* lb = (v4u*)&Bs[0][srow][scol];
    #pragma unroll
    for (int i = 0; i < 4; ++i) { la[i] = ga[i]; lb[i] = gb[i]; }
  }
  __syncthreads();

  const int niter = K >> 6;
  for (int it = 0; it < niter; ++it) {
    const int cur = it & 1;
    const bool more = (it + 1 < niter);

    // issue next tile's global loads first (latency hidden under WMMA)
    v4u ra[4], rb[4];
    if (more) {
      const v4u* ga = (const v4u*)(gA + (size_t)(it + 1) * 64);
      const v4u* gb = (const v4u*)(gB + (size_t)(it + 1) * 64);
      #pragma unroll
      for (int i = 0; i < 4; ++i) { ra[i] = ga[i]; rb[i] = gb[i]; }
    }

    // compute current tile out of buffer `cur`
    #pragma unroll
    for (int kk = 0; kk < 64; kk += 32) {
      v16bf af[2], bfr[4];
      #pragma unroll
      for (int mi = 0; mi < 2; ++mi) {
        int r = wm * 32 + mi * 16 + (lane & 15);
        FragCvt f;
        f.u2[0] = *(const v4u*)&As[cur][r][kk + kbA];
        f.u2[1] = *(const v4u*)&As[cur][r][kk + kbA + 16];
        af[mi] = f.b;
      }
      #pragma unroll
      for (int ni = 0; ni < 4; ++ni) {
        int cc = wn * 64 + ni * 16 + (lane & 15);
        FragCvt f;
        f.u2[0] = *(const v4u*)&Bs[cur][cc][kk + kbA];
        f.u2[1] = *(const v4u*)&Bs[cur][cc][kk + kbA + 16];
        bfr[ni] = f.b;
      }
      #pragma unroll
      for (int mi = 0; mi < 2; ++mi) {
        #pragma unroll
        for (int ni = 0; ni < 4; ++ni) {
          acc[mi][ni] = __builtin_amdgcn_wmma_f32_16x16x32_bf16(
              false, af[mi], false, bfr[ni], (short)0, acc[mi][ni], false, false);
        }
      }
    }

    // store next tile into the other buffer
    if (more) {
      v4u* la = (v4u*)&As[cur ^ 1][srow][scol];
      v4u* lb = (v4u*)&Bs[cur ^ 1][srow][scol];
      #pragma unroll
      for (int i = 0; i < 4; ++i) { la[i] = ra[i]; lb[i] = rb[i]; }
    }
    __syncthreads();
  }

  if (out_mode == 0) {
    float* C = (float*)Cout;
    #pragma unroll
    for (int mi = 0; mi < 2; ++mi) {
      #pragma unroll
      for (int ni = 0; ni < 4; ++ni) {
        #pragma unroll
        for (int r = 0; r < 8; ++r) {
          int row = m0 + wm * 32 + mi * 16 + r + hi8;
          int col = n0 + wn * 64 + ni * 16 + (lane & 15);
          C[(size_t)row * D_DIM + col] = acc[mi][ni][r];
        }
      }
    }
  } else {
    // bf16 transposed output Ct[col][row], rows contiguous -> packed b128
    unsigned short* Ct = (unsigned short*)Cout;
    #pragma unroll
    for (int mi = 0; mi < 2; ++mi) {
      #pragma unroll
      for (int ni = 0; ni < 4; ++ni) {
        int col  = n0 + wn * 64 + ni * 16 + (lane & 15);
        int rowb = m0 + wm * 32 + mi * 16 + hi8;
        PackBF8 p;
        #pragma unroll
        for (int r = 0; r < 8; ++r) p.s[r] = f2bfu(acc[mi][ni][r]);
        *(v4u*)&Ct[(size_t)col * M + rowb] = p.u;
      }
    }
  }
}

// ---------------------------------------------------------------------------
// RMS norm over D=1024 (one block per row), multi-format outputs:
//   Yf  (f32, optional, may alias X)        Res (f32 residual, optional)
//   Ybf (bf16, optional)                    Ype (bf16 + pe(row&4095), optional)
// ---------------------------------------------------------------------------
__global__ __launch_bounds__(256)
void rmsnorm_kernel(const float* __restrict__ X, const float* __restrict__ Res,
                    const float* __restrict__ w, float* __restrict__ Yf,
                    unsigned short* __restrict__ Ybf, unsigned short* __restrict__ Ype) {
  __shared__ float red[256];
  const int row = blockIdx.x, tid = threadIdx.x;
  float x[4];
  float ss = 0.f;
  #pragma unroll
  for (int i = 0; i < 4; ++i) {
    int d = tid * 4 + i;
    float v = X[(size_t)row * D_DIM + d];
    if (Res) v += Res[(size_t)row * D_DIM + d];
    x[i] = v;
    ss += v * v;
  }
  red[tid] = ss;
  __syncthreads();
  for (int s = 128; s > 0; s >>= 1) {
    if (tid < s) red[tid] += red[tid + s];
    __syncthreads();
  }
  const float scale = rsqrtf(red[0] * (1.0f / (float)D_DIM) + RMS_EPS);
  const int srow = row & (S_LEN - 1);
  #pragma unroll
  for (int i = 0; i < 4; ++i) {
    int d = tid * 4 + i;
    float y = x[i] * scale * w[d];
    if (Yf)  Yf[(size_t)row * D_DIM + d]  = y;
    if (Ybf) Ybf[(size_t)row * D_DIM + d] = f2bfu(y);
    if (Ype) Ype[(size_t)row * D_DIM + d] = f2bfu(y + pe_val(srow, d));
  }
}

// ---------------------------------------------------------------------------
// Gather q_states rows; emit qst (f32 residual), qin_bf (= qst + pe(pos)),
// and clamped positions. grid = B*N blocks.
// ---------------------------------------------------------------------------
__global__ __launch_bounds__(256)
void gather_kernel(const float* __restrict__ fused, const int* __restrict__ ctx,
                   float* __restrict__ qst, unsigned short* __restrict__ qin_bf,
                   int* __restrict__ posb) {
  const int bn = blockIdx.x;
  const int b  = bn >> 9;           // N=512
  int p = ctx[bn];
  p = (p < 0) ? 0 : ((p > S_LEN - 1) ? (S_LEN - 1) : p);
  if (threadIdx.x == 0) posb[bn] = p;
  const size_t src = ((size_t)b * S_LEN + p) * D_DIM;
  const size_t dst = (size_t)bn * D_DIM;
  #pragma unroll
  for (int i = 0; i < 4; ++i) {
    int d = threadIdx.x * 4 + i;
    float v = fused[src + d];
    qst[dst + d] = v;
    qin_bf[dst + d] = f2bfu(v + pe_val(p, d));
  }
}

// ---------------------------------------------------------------------------
// Flash attention: one block per 16 queries (grid = B*N/16 = 64), 8 waves.
// Q,K bf16 row-major [*,1024]; V bf16 transposed [1024, 8192].
// Wave w owns feature slice [w*128, w*128+128).
// Output: attn bf16 row-major [B*N, 1024].
// ---------------------------------------------------------------------------
__global__ __launch_bounds__(256)
void attention_kernel(const unsigned short* __restrict__ Q,
                      const unsigned short* __restrict__ Kg,
                      const unsigned short* __restrict__ Vt,
                      const int* __restrict__ posb,
                      unsigned short* __restrict__ Oout) {
  __shared__ __align__(16) unsigned short Qs[16][1024];   // 32 KB
  __shared__ float sPart[8][16][32];                      // 16 KB
  __shared__ __align__(16) unsigned short Pbf[16][32];    // 1 KB
  __shared__ float mrow[16], lrow[16], arow_[16];
  __shared__ int qpos[16];
  __shared__ int tmaxSh;

  const int tid  = threadIdx.x;
  const int lane = tid & 31;
  const int wid  = tid >> 5;
  const int b    = blockIdx.x >> 5;
  const int qb   = blockIdx.x & 31;
  const int n0   = qb * 16;
  const int rowBase = b * N_CTX + n0;
  const int ds   = wid * 128;

  // stage Q tile (bf16, b128 copies)
  {
    int r  = tid >> 4;
    int cb = (tid & 15) * 64;
    const v4u* src = (const v4u*)(Q + (size_t)(rowBase + r) * D_DIM + cb);
    v4u* dst = (v4u*)&Qs[r][cb];
    #pragma unroll
    for (int i = 0; i < 8; ++i) dst[i] = src[i];
  }
  if (tid < 16) {
    qpos[tid] = posb[rowBase + tid];
    mrow[tid] = -3.0e38f;
    lrow[tid] = 0.f;
  }
  __syncthreads();
  if (tid == 0) {
    int mx = 0;
    for (int i = 0; i < 16; ++i) mx = (qpos[i] > mx) ? qpos[i] : mx;
    tmaxSh = mx + 1;
  }
  __syncthreads();
  const int tmax = tmaxSh;

  const int kbA = (lane < 16) ? 0 : 8;
  const int kbB = (lane < 16) ? 0 : 16;
  const int hi8 = (lane < 16) ? 0 : 8;

  // preload Q fragments (4 chunks of K=32, b128 pairs at k offsets kbA and kbA+16)
  v16bf qf[4];
  {
    const int r = lane & 15;
    #pragma unroll
    for (int c = 0; c < 4; ++c) {
      FragCvt f;
      f.u2[0] = *(const v4u*)&Qs[r][ds + c * 32 + kbA];
      f.u2[1] = *(const v4u*)&Qs[r][ds + c * 32 + kbA + 16];
      qf[c] = f.b;
    }
  }

  v8f vzero = {0.f,0.f,0.f,0.f,0.f,0.f,0.f,0.f};
  v8f oacc[8];
  #pragma unroll
  for (int i = 0; i < 8; ++i) oacc[i] = vzero;

  const float scale = 0.03125f;   // 1024^-0.5

  for (int t0 = 0; t0 < tmax; t0 += 32) {
    // ---- partial scores over this wave's 128 features ----
    v8f s0 = vzero, s1 = vzero;
    #pragma unroll
    for (int c = 0; c < 4; ++c) {
      // K row-major: per-lane two 16B chunks at k offsets kbA, kbA+16
      const unsigned short* k0p =
          Kg + ((size_t)b * S_LEN + t0 + (lane & 15)) * D_DIM + ds + c * 32 + kbA;
      const unsigned short* k1p = k0p + (size_t)16 * D_DIM;
      FragCvt f0, f1;
      f0.u2[0] = *(const v4u*)k0p;  f0.u2[1] = *(const v4u*)(k0p + 16);
      f1.u2[0] = *(const v4u*)k1p;  f1.u2[1] = *(const v4u*)(k1p + 16);
      s0 = __builtin_amdgcn_wmma_f32_16x16x32_bf16(false, qf[c], false, f0.b, (short)0, s0, false, false);
      s1 = __builtin_amdgcn_wmma_f32_16x16x32_bf16(false, qf[c], false, f1.b, (short)0, s1, false, false);
    }
    #pragma unroll
    for (int r = 0; r < 8; ++r) {
      sPart[wid][r + hi8][lane & 15]        = s0[r];
      sPart[wid][r + hi8][16 + (lane & 15)] = s1[r];
    }
    __syncthreads();

    // ---- online softmax (threads 0..15, one query row each) ----
    if (tid < 16) {
      const int r = tid;
      float mold = mrow[r], mx = mold;
      float sv[32];
      for (int c = 0; c < 32; ++c) {
        float s = 0.f;
        for (int w8 = 0; w8 < 8; ++w8) s += sPart[w8][r][c];
        s *= scale;
        if (t0 + c > qpos[r]) s = -3.0e38f;   // mask: keep t <= pos
        sv[c] = s;
        mx = fmaxf(mx, s);
      }
      float alpha = expf(mold - mx);
      float l = lrow[r] * alpha;
      for (int c = 0; c < 32; ++c) {
        float p = expf(sv[c] - mx);
        l += p;
        Pbf[r][c] = f2bfu(p);
      }
      mrow[r] = mx; lrow[r] = l; arow_[r] = alpha;
    }
    __syncthreads();

    // ---- rescale accumulators, then PV ----
    float al[8];
    #pragma unroll
    for (int r = 0; r < 8; ++r) al[r] = arow_[r + hi8];
    #pragma unroll
    for (int nt = 0; nt < 8; ++nt) {
      #pragma unroll
      for (int r = 0; r < 8; ++r) oacc[nt][r] *= al[r];
    }
    v16bf pf;
    {
      const int r = lane & 15;
      FragCvt f;
      f.u2[0] = *(const v4u*)&Pbf[r][kbA];
      f.u2[1] = *(const v4u*)&Pbf[r][kbA + 16];
      pf = f.b;
    }
    #pragma unroll
    for (int nt = 0; nt < 8; ++nt) {
      // V transposed [1024, 8192]: per-lane contiguous over key index
      const unsigned short* vp =
          Vt + ((size_t)(ds + nt * 16 + (lane & 15))) * (size_t)(2 * S_LEN)
             + (size_t)b * S_LEN + t0 + kbA;
      FragCvt f;
      f.u2[0] = *(const v4u*)vp;  f.u2[1] = *(const v4u*)(vp + 16);
      oacc[nt] = __builtin_amdgcn_wmma_f32_16x16x32_bf16(false, pf, false, f.b, (short)0, oacc[nt], false, false);
    }
    __syncthreads();
  }

  // ---- epilogue: divide by l, store bf16 row-major ----
  float linv[8];
  #pragma unroll
  for (int r = 0; r < 8; ++r) {
    float l = lrow[r + hi8];
    linv[r] = (l > 0.f) ? (1.f / l) : 0.f;
  }
  #pragma unroll
  for (int nt = 0; nt < 8; ++nt) {
    #pragma unroll
    for (int r = 0; r < 8; ++r) {
      int row = rowBase + r + hi8;
      int col = ds + nt * 16 + (lane & 15);
      Oout[(size_t)row * D_DIM + col] = f2bfu(oacc[nt][r] * linv[r]);
    }
  }
}

// ---------------------------------------------------------------------------
// Host launcher
// ---------------------------------------------------------------------------
extern "C" void kernel_launch(void* const* d_in, const int* in_sizes, int n_in,
                              void* d_out, int out_size, void* d_ws, size_t ws_size,
                              hipStream_t stream) {
  (void)in_sizes; (void)n_in; (void)out_size; (void)ws_size;
  const float* hidden = (const float*)d_in[0];   // [8192, 3072]
  const int*   ctx    = (const int*)  d_in[1];   // [1024]
  const float* Wfc    = (const float*)d_in[2];   // [3072,1024]
  const float* w_h    = (const float*)d_in[3];
  const float* Wq     = (const float*)d_in[4];
  const float* Wk     = (const float*)d_in[5];
  const float* Wv     = (const float*)d_in[6];
  const float* Wo     = (const float*)d_in[7];
  const float* w_q    = (const float*)d_in[8];
  const float* w_k    = (const float*)d_in[9];
  const float* w_o    = (const float*)d_in[10];
  float* out = (float*)d_out;

  char* wp = (char*)d_ws;
  auto alloc = [&](size_t bytes) { char* p = wp; wp += (bytes + 255) & ~(size_t)255; return p; };

  unsigned short* hidden_bf = (unsigned short*)alloc((size_t)8192 * 3072 * 2); // 50.3MB (reused for k_raw/k_bf)
  unsigned short* wfc_t = (unsigned short*)alloc((size_t)3072 * 1024 * 2);
  unsigned short* wq_t  = (unsigned short*)alloc((size_t)1024 * 1024 * 2);
  unsigned short* wk_t  = (unsigned short*)alloc((size_t)1024 * 1024 * 2);
  unsigned short* wv_t  = (unsigned short*)alloc((size_t)1024 * 1024 * 2);
  unsigned short* wo_t  = (unsigned short*)alloc((size_t)1024 * 1024 * 2);
  float*          fused = (float*)alloc((size_t)8192 * 1024 * 4);              // 33.5MB (reused for v_t)
  unsigned short* fused_bf = (unsigned short*)alloc((size_t)8192 * 1024 * 2);
  unsigned short* hist_bf  = (unsigned short*)alloc((size_t)8192 * 1024 * 2);
  float*          qst   = (float*)alloc((size_t)1024 * 1024 * 4);
  unsigned short* qin_bf = (unsigned short*)alloc((size_t)1024 * 1024 * 2);
  float*          qraw  = (float*)alloc((size_t)1024 * 1024 * 4);
  unsigned short* q_bf  = (unsigned short*)alloc((size_t)1024 * 1024 * 2);
  unsigned short* attn_bf = (unsigned short*)alloc((size_t)1024 * 1024 * 2);
  float*          oraw  = (float*)alloc((size_t)1024 * 1024 * 4);
  int*            posb  = (int*)alloc(1024 * 4);

  // region reuse (sequenced by stream order):
  float*          k_raw = (float*)hidden_bf;                                   // after GEMM1
  unsigned short* k_bf  = (unsigned short*)((char*)hidden_bf + (size_t)8192 * 1024 * 4);
  unsigned short* v_t   = (unsigned short*)fused;                              // after gather

  dim3 thr(256);

  // one-time bf16 conversion / weight transposition
  conv_bf16_kernel<<<24576, thr, 0, stream>>>(hidden, hidden_bf);              // 8192*3072
  transpose_conv_kernel<<<dim3(32, 96), thr, 0, stream>>>(Wfc, wfc_t, 3072, 1024);
  transpose_conv_kernel<<<dim3(32, 32), thr, 0, stream>>>(Wq, wq_t, 1024, 1024);
  transpose_conv_kernel<<<dim3(32, 32), thr, 0, stream>>>(Wk, wk_t, 1024, 1024);
  transpose_conv_kernel<<<dim3(32, 32), thr, 0, stream>>>(Wv, wv_t, 1024, 1024);
  transpose_conv_kernel<<<dim3(32, 32), thr, 0, stream>>>(Wo, wo_t, 1024, 1024);

  // fused = rms_norm(concat(hidden) @ W_fc); also emit bf16 + (bf16 + pe(s))
  gemm_bf16_kernel<<<dim3(8, 64), thr, 0, stream>>>(hidden_bf, wfc_t, fused, 8192, 3072, 0);
  rmsnorm_kernel<<<8192, thr, 0, stream>>>(fused, nullptr, w_h, fused, fused_bf, hist_bf);

  // q_states gather (+ pe(pos) bf16 input for Wq)
  gather_kernel<<<1024, thr, 0, stream>>>(fused, ctx, qst, qin_bf, posb);

  // k = rms_norm(hist @ Wk); v^T = (fused @ Wv)^T; q = rms_norm(qin @ Wq)
  gemm_bf16_kernel<<<dim3(8, 64), thr, 0, stream>>>(hist_bf, wk_t, k_raw, 8192, 1024, 0);
  rmsnorm_kernel<<<8192, thr, 0, stream>>>(k_raw, nullptr, w_k, nullptr, k_bf, nullptr);
  gemm_bf16_kernel<<<dim3(8, 64), thr, 0, stream>>>(fused_bf, wv_t, v_t, 8192, 1024, 1);
  gemm_bf16_kernel<<<dim3(8, 8),  thr, 0, stream>>>(qin_bf, wq_t, qraw, 1024, 1024, 0);
  rmsnorm_kernel<<<1024, thr, 0, stream>>>(qraw, nullptr, w_q, nullptr, q_bf, nullptr);

  // masked flash attention
  attention_kernel<<<64, thr, 0, stream>>>(q_bf, k_bf, v_t, posb, attn_bf);

  // out = rms_norm(q_states + attn @ Wo)
  gemm_bf16_kernel<<<dim3(8, 8), thr, 0, stream>>>(attn_bf, wo_t, oraw, 1024, 1024, 0);
  rmsnorm_kernel<<<1024, thr, 0, stream>>>(oraw, qst, w_o, out, nullptr, nullptr);
}
